// SpNet_68298569941096
// MI455X (gfx1250) — matrix operand order repaired and checked by reading, not compiled, not run on hardware
//
#include <hip/hip_runtime.h>

typedef __attribute__((ext_vector_type(16))) _Float16 v16h;
typedef __attribute__((ext_vector_type(8)))  _Float16 v8h;
typedef __attribute__((ext_vector_type(8)))  float    v8f;

#define N_PTS 8192
#define P_TOT 65536     // K(=8) * N_PTS
#define EPS_F 1e-5f
#define KNN_TILE 2048

// ---------------------------------------------------------------- fc: h = x @ W + b
__global__ void fc_kernel(const float* __restrict__ x, const float* __restrict__ W,
                          const float* __restrict__ b, float* __restrict__ h) {
    int t = blockIdx.x * blockDim.x + threadIdx.x;     // n*32 + c
    if (t >= N_PTS * 32) return;
    int n = t >> 5, c = t & 31;
    float acc = b[c];
#pragma unroll
    for (int d = 0; d < 5; ++d) acc += x[n * 5 + d] * W[d * 32 + c];
    h[t] = acc;
}

// ---------------------------------------------------------------- kNN: one wave per point
__global__ __launch_bounds__(256)
void knn_kernel(const float* __restrict__ x, int* __restrict__ idx) {
    __shared__ float xs[KNN_TILE * 5];                 // 40 KB tile of points
    const int lane = threadIdx.x & 31;
    const int wave = threadIdx.x >> 5;
    const int i    = blockIdx.x * 8 + wave;            // point owned by this wave

    float xi[5], sqi = 0.f;
#pragma unroll
    for (int d = 0; d < 5; ++d) { xi[d] = x[i * 5 + d]; sqi += xi[d] * xi[d]; }

    float bd[7]; int bi[7];
#pragma unroll
    for (int m = 0; m < 7; ++m) { bd[m] = 3.4e38f; bi[m] = 0x7fffffff; }

    for (int tile = 0; tile < N_PTS; tile += KNN_TILE) {
        __syncthreads();
        for (int t = threadIdx.x; t < KNN_TILE * 5; t += 256)
            xs[t] = x[tile * 5 + t];
        __syncthreads();
        for (int jj = lane; jj < KNN_TILE; jj += 32) {
            const float* xj = xs + jj * 5;
            float dot = 0.f, sqj = 0.f;
#pragma unroll
            for (int d = 0; d < 5; ++d) { float v = xj[d]; dot += xi[d] * v; sqj += v * v; }
            float dist = sqi + sqj - 2.f * dot;
            if (dist < bd[6]) {                         // strict < : stable like top_k
                int j = tile + jj;
                int pos = 7;
                while (pos > 0 && dist < bd[pos - 1]) --pos;
#pragma unroll
                for (int s = 6; s > 0; --s)
                    if (s > pos) { bd[s] = bd[s - 1]; bi[s] = bi[s - 1]; }
                bd[pos] = dist; bi[pos] = j;
            }
        }
    }

    // merge 32 sorted per-lane lists: 7 rounds of wave-wide (d, idx) argmin
#pragma unroll
    for (int m = 0; m < 7; ++m) {
        float cd = bd[0]; int cidx = bi[0];
        float md = cd; int mi = cidx;
#pragma unroll
        for (int off = 16; off > 0; off >>= 1) {
            float od = __shfl_xor(md, off, 32);
            int   oi = __shfl_xor(mi, off, 32);
            if (od < md || (od == md && oi < mi)) { md = od; mi = oi; }
        }
        if (cd == md && cidx == mi) {                  // unique winner pops its head
#pragma unroll
            for (int s = 0; s < 6; ++s) { bd[s] = bd[s + 1]; bi[s] = bi[s + 1]; }
            bd[6] = 3.4e38f; bi[6] = 0x7fffffff;
        }
        if (lane == 0) idx[i * 7 + m] = mi;
    }
}

// ---------------------------------------------------------------- gather -> f32 [c][p] + f16 [p][c]
__global__ void gather_kernel(const float* __restrict__ h, const int* __restrict__ idx,
                              float* __restrict__ t0, _Float16* __restrict__ t16) {
    int t = blockIdx.x * blockDim.x + threadIdx.x;     // c*P_TOT + k*N_PTS + n
    if (t >= 32 * P_TOT) return;
    int n = t & (N_PTS - 1);
    int k = (t >> 13) & 7;
    int c = t >> 16;
    int p = t & (P_TOT - 1);
    int src = (k == 0) ? n : idx[n * 7 + (k - 1)];
    float v = h[src * 32 + c];
    t0[t] = v;
    t16[(size_t)p * 32 + c] = (_Float16)v;
}

// ---------------------------------------------------------------- weight/bias prep (pad + f16 repack)
// wf layout: [(d*NCH + ch)*co_pad + row][32 channels-in-chunk], rows >= co zeroed.
__global__ void prep_kernel(const float* __restrict__ w, const float* __restrict__ bias,
                            _Float16* __restrict__ wf, float* __restrict__ biasp,
                            int ci, int co, int co_pad) {
    int nch = ci >> 5;
    int total = 3 * nch * co_pad * 32;
    int t = blockIdx.x * blockDim.x + threadIdx.x;
    if (t >= total) return;
    int kk  = t & 31;
    int row = (t >> 5) % co_pad;
    int fr  = t / (32 * co_pad);           // d*nch + ch
    int ch  = fr % nch;
    int d   = fr / nch;
    float v = (row < co) ? w[((size_t)row * ci + (ch * 32 + kk)) * 3 + d] : 0.f;
    wf[t] = (_Float16)v;
    if (t < co_pad) biasp[t] = (t < co) ? bias[t] : 0.f;
}

// ---------------------------------------------------------------- conv (3,1) as WMMA GEMM
// One wave computes a 16(co) x 64(p) strip: 4 accumulators, A fragments reused 4x.
template <int CI>
__global__ __launch_bounds__(256)
void conv_wmma_kernel(const _Float16* __restrict__ t16, const _Float16* __restrict__ wf,
                      const float* __restrict__ biasp, float* __restrict__ z, int co_pad) {
    constexpr int NCH = CI / 32;
    const int lane   = threadIdx.x & 31;
    const int wave   = threadIdx.x >> 5;
    const int pg     = blockIdx.x * 8 + wave;          // 1024 p-groups of 64
    const int p0     = pg * 64;
    const int k      = p0 >> 13;                       // constant within the 64-wide group
    const int n0     = p0 & (N_PTS - 1);
    const int co0    = blockIdx.y * 16;
    const int laneHi = lane >> 4;                      // 0: lanes 0-15, 1: lanes 16-31
    const int lm     = lane & 15;

    // bias init (padded -> unconditional); C/D layout: row = r + 8*laneHi, col = lane&15
    const float4* b4 = (const float4*)(biasp + co0 + 8 * laneHi);
    float4 blo = b4[0], bhi = b4[1];
    v8f acc[4];
#pragma unroll
    for (int s = 0; s < 4; ++s) {
        acc[s][0] = blo.x; acc[s][1] = blo.y; acc[s][2] = blo.z; acc[s][3] = blo.w;
        acc[s][4] = bhi.x; acc[s][5] = bhi.y; acc[s][6] = bhi.z; acc[s][7] = bhi.w;
    }

#pragma unroll
    for (int d = 0; d < 3; ++d) {
        const int koff = k + d - 1;
        if (koff < 0 || koff >= 8) continue;           // zero-pad taps (wave-uniform)
        const _Float16* brow = t16 + (size_t)(koff * N_PTS + n0 + lm) * CI;
#pragma unroll
        for (int ch = 0; ch < NCH; ++ch) {
            // A 16x32 f16 fragment: M = lane&15, K = 16*(j>=8) + 8*laneHi + (j&7)
            const _Float16* abase = wf + ((size_t)(d * NCH + ch) * co_pad + (co0 + lm)) * 32;
            v8h alo = *(const v8h*)(abase + 8 * laneHi);
            v8h ahi = *(const v8h*)(abase + 16 + 8 * laneHi);
            v16h a = __builtin_shufflevector(alo, ahi, 0, 1, 2, 3, 4, 5, 6, 7,
                                             8, 9, 10, 11, 12, 13, 14, 15);
#pragma unroll
            for (int s = 0; s < 4; ++s) {
                // B 32x16 f16 fragment: N = lane&15, K = 16*(j>=8) + 8*laneHi + (j&7) (channel-fastest t16)
                const _Float16* bb = brow + (size_t)(16 * s) * CI + ch * 32 + 8 * laneHi;
                v8h q0 = *(const v8h*)(bb);
                v8h q1 = *(const v8h*)(bb + 16);
                v16h b = __builtin_shufflevector(q0, q1, 0, 1, 2, 3, 4, 5, 6, 7,
                                                 8, 9, 10, 11, 12, 13, 14, 15);
                acc[s] = __builtin_amdgcn_wmma_f32_16x16x32_f16(
                    false, a, false, b, (short)0, acc[s], false, false);
            }
        }
    }

#pragma unroll
    for (int s = 0; s < 4; ++s)
#pragma unroll
        for (int r = 0; r < 8; ++r)
            z[(size_t)(co0 + r + 8 * laneHi) * P_TOT + p0 + 16 * s + lm] = acc[s][r];
    (void)co_pad;
}

// ---------------------------------------------------------------- per-channel stats (folded double-norm)
__global__ __launch_bounds__(256)
void stats_kernel(const float* __restrict__ z, float* __restrict__ stats) {
    int c = blockIdx.x;
    const float4* p4 = (const float4*)(z + (size_t)c * P_TOT);
    float s = 0.f, s2 = 0.f;
    for (int i = threadIdx.x; i < P_TOT / 4; i += 256) {
        float4 v = p4[i];
        s  += v.x + v.y + v.z + v.w;
        s2 += v.x * v.x + v.y * v.y + v.z * v.z + v.w * v.w;
    }
    __shared__ float sh[256], sh2[256];
    sh[threadIdx.x] = s; sh2[threadIdx.x] = s2;
    __syncthreads();
    for (int off = 128; off > 0; off >>= 1) {
        if (threadIdx.x < off) { sh[threadIdx.x] += sh[threadIdx.x + off];
                                 sh2[threadIdx.x] += sh2[threadIdx.x + off]; }
        __syncthreads();
    }
    if (threadIdx.x == 0) {
        float mean = sh[0] * (1.f / P_TOT);
        float var  = sh2[0] * (1.f / P_TOT) - mean * mean;
        if (var < 0.f) var = 0.f;
        float r1 = rsqrtf(var + EPS_F);                // norm #1
        float v2 = var / (var + EPS_F);                // var after norm #1
        float scale = r1 * rsqrtf(v2 + EPS_F);         // norm #2 folded in
        stats[2 * c] = mean; stats[2 * c + 1] = scale;
    }
}

// ---------------------------------------------------------------- affine + residual, emit f32 + f16
__global__ void apply_kernel(const float* __restrict__ z, const float* __restrict__ stats,
                             const float* __restrict__ g, const float* __restrict__ be,
                             const float* __restrict__ res, float* __restrict__ out,
                             _Float16* __restrict__ t16,
                             int total, int co, int use_res) {
    int t = blockIdx.x * blockDim.x + threadIdx.x;
    if (t >= total) return;
    int c = t >> 16;                                    // t / P_TOT
    int p = t & (P_TOT - 1);
    float v = (z[t] - stats[2 * c]) * stats[2 * c + 1];
    v = g[c] * v + be[c];
    if (use_res) v += res[t];
    out[t] = v;
    t16[(size_t)p * co + c] = (_Float16)v;              // next block's WMMA input
}

// ----------------------------------------------------------------
extern "C" void kernel_launch(void* const* d_in, const int* in_sizes, int n_in,
                              void* d_out, int out_size, void* d_ws, size_t ws_size,
                              hipStream_t stream) {
    (void)in_sizes; (void)n_in; (void)out_size; (void)ws_size;
    const float* x    = (const float*)d_in[0];
    const float* W_fc = (const float*)d_in[1];
    const float* b_fc = (const float*)d_in[2];
    const float* cw[7], *cb[7], *gg[7], *bb[7];
    for (int i = 0; i < 7; ++i) {
        cw[i] = (const float*)d_in[3 + i];
        cb[i] = (const float*)d_in[10 + i];
        gg[i] = (const float*)d_in[17 + i];
        bb[i] = (const float*)d_in[24 + i];
    }

    char*      ws    = (char*)d_ws;
    float*     bufA  = (float*)ws;                                 // 64 MB f32 activations
    float*     bufB  = (float*)(ws + ((size_t)64 << 20));          // 64 MB f32 conv out
    _Float16*  t16   = (_Float16*)(ws + ((size_t)128 << 20));      // 32 MB f16 activations [p][c]
    float*     h     = (float*)(ws + ((size_t)160 << 20));         // 1 MB
    int*       idx   = (int*)  (ws + ((size_t)161 << 20));         // 224 KB
    float*     stats = (float*)(ws + ((size_t)162 << 20));         // 2 KB
    _Float16*  wf    = (_Float16*)(ws + ((size_t)163 << 20));      // 192 KB padded f16 weights
    float*     biasp = (float*)(ws + ((size_t)164 << 20));         // 1 KB padded bias

    fc_kernel    <<<(N_PTS * 32 + 255) / 256, 256, 0, stream>>>(x, W_fc, b_fc, h);
    knn_kernel   <<<N_PTS / 8,                256, 0, stream>>>(x, idx);   // 1 wave / point
    gather_kernel<<<(32 * P_TOT + 255) / 256, 256, 0, stream>>>(h, idx, bufA, t16);

    const int INS[7]  = {32, 32, 64, 64, 128, 128, 256};
    const int OUTS[7] = {32, 64, 64, 128, 128, 256, 1};
    float* tin = bufA; float* tout = bufB;
    for (int i = 0; i < 7; ++i) {
        int ci = INS[i], co = OUTS[i];
        int co_pad = ((co + 15) / 16) * 16;
        int ptot = 3 * (ci / 32) * co_pad * 32;
        prep_kernel<<<(ptot + 255) / 256, 256, 0, stream>>>(cw[i], cb[i], wf, biasp,
                                                            ci, co, co_pad);
        dim3 grid(128, co_pad / 16);                   // 1024 p-groups / 8 waves per block
        switch (ci) {
            case 32:  conv_wmma_kernel<32> <<<grid, 256, 0, stream>>>(t16, wf, biasp, tout, co_pad); break;
            case 64:  conv_wmma_kernel<64> <<<grid, 256, 0, stream>>>(t16, wf, biasp, tout, co_pad); break;
            case 128: conv_wmma_kernel<128><<<grid, 256, 0, stream>>>(t16, wf, biasp, tout, co_pad); break;
            case 256: conv_wmma_kernel<256><<<grid, 256, 0, stream>>>(t16, wf, biasp, tout, co_pad); break;
        }
        stats_kernel<<<co, 256, 0, stream>>>(tout, stats);
        int total = co * P_TOT;
        float* dst = (i == 6) ? (float*)d_out : tout;
        apply_kernel<<<(total + 255) / 256, 256, 0, stream>>>(
            tout, stats, gg[i], bb[i], tin, dst, t16, total, co, ci == co ? 1 : 0);
        float* tmp = tin; tin = tout; tout = tmp;
    }
}
